// TemporalAttentionBilinearLayer_84112639525714
// MI455X (gfx1250) — matrix operand order repaired and checked by reading, not compiled, not run on hardware
//
#include <hip/hip_runtime.h>
#include <hip/hip_bf16.h>
#include <math.h>

// ---------------- types ----------------
typedef __attribute__((ext_vector_type(16))) __bf16 v16bf;
typedef __attribute__((ext_vector_type(8)))  float  v8f;

#define B_SZ   512
#define D1_IN  40
#define T_DIM  1024
#define D1_OUT 120
#define D2_OUT 64

__device__ inline unsigned pack_bf16(float a, float b) {
  union { __bf16 h[2]; unsigned u; } u_;
  u_.h[0] = (__bf16)a;
  u_.h[1] = (__bf16)b;
  return u_.u;
}

__device__ inline v8f zero8() {
  v8f z;
#pragma unroll
  for (int i = 0; i < 8; ++i) z[i] = 0.0f;
  return z;
}

__device__ inline v8f wmma_bf16(v16bf a, v16bf b, v8f c) {
  // (neg_a, A, neg_b, B, c_mod, C, reuse_a, reuse_b)
  return __builtin_amdgcn_wmma_f32_16x16x32_bf16(false, a, false, b, (short)0, c,
                                                 false, false);
}

// Fragment = 16 contiguous bf16 (32 bytes) in LDS, stored in contiguous k order at staging time.
__device__ inline v16bf frag_at(const unsigned* p) { return *(const v16bf*)p; }

// swizzled position of k (0..31) within an A-operand row chunk:
// lanes 0-15 need K {0..7,16..23} first, lanes 16-31 need K {8..15,24..31}.
__device__ inline int a_swz32(int k) {
  return ((k & 8) << 1) + ((k & 16) >> 1) + (k & 7);
}

__device__ inline float redmax16(float v) {
#pragma unroll
  for (int m = 8; m >= 1; m >>= 1) v = fmaxf(v, __shfl_xor(v, m, 32));
  return v;
}
__device__ inline float redsum16(float v) {
#pragma unroll
  for (int m = 8; m >= 1; m >>= 1) v += __shfl_xor(v, m, 32);
  return v;
}

// ====================================================================
// Kernel 1: E = x @ W (M = 20480, K = N = 1024), fused row softmax -> A.
// Block = 16 rows x full N=1024. Wave w owns cols {w*64..}+{512+w*64..}.
// LDS tiles stored in fragment order:
//   Atile32[m*16 + d]    : dword d = bf16 pair (k=2*d, 2*d+1) of row m
//   Btile32[n*16 + k2]   : dword = bf16 pair (k=2*k2, 2*k2+1) of column n
// ====================================================================
__global__ __launch_bounds__(256) void k_attn_softmax(const float* __restrict__ x,
                                                      const float* __restrict__ W,
                                                      float* __restrict__ Aout) {
  __shared__ __align__(32) unsigned Atile32[16 * 16];    // 1 KB
  __shared__ __align__(32) unsigned Btile32[512 * 16];   // 32 KB (half of N)
  __shared__ float red[8 * 16];
  __shared__ float rowstat[16];

  const int tid    = threadIdx.x;
  const int wave   = tid >> 5;
  const int lane   = tid & 31;
  const int laneLo = lane & 15;
  const int laneHi = lane >> 4;

  const int row0 = blockIdx.x * 16;
  const float* xrow = x + (size_t)row0 * T_DIM;

  // A-tile staging assignment: one bf16 pair per thread (16 rows x 16 pairs)
  const int m_st = tid >> 4;
  const int k_st = (tid & 15) * 2;
  const int dA   = k_st >> 1;            // dword slot within row

  v8f acc[8];
#pragma unroll
  for (int t = 0; t < 8; ++t) acc[t] = zero8();

  for (int k0 = 0; k0 < T_DIM; k0 += 32) {
    // ---- stage 16x32 A tile: one float2 load + one packed dword store ----
    {
      const float2 g = *(const float2*)(xrow + (size_t)m_st * T_DIM + k0 + k_st);
      Atile32[m_st * 16 + dA] = pack_bf16(g.x, g.y);
    }
    v16bf afrag;
#pragma unroll
    for (int h = 0; h < 2; ++h) {
      __syncthreads();
      // ---- stage 32 x 512 half of W rows [k0,k0+32), transposed [n][k] ----
#pragma unroll
      for (int it = 0; it < 8; ++it) {
        int i  = it * 256 + tid;
        int r2 = i >> 7;            // k-pair 0..15
        int n4 = i & 127;           // group of 4 columns
        const float* wp = W + (size_t)(k0 + 2 * r2) * T_DIM + h * 512 + n4 * 4;
        const float4 g0 = *(const float4*)(wp);
        const float4 g1 = *(const float4*)(wp + T_DIM);
        int nb = n4 * 4;
        Btile32[(nb + 0) * 16 + r2] = pack_bf16(g0.x, g1.x);
        Btile32[(nb + 1) * 16 + r2] = pack_bf16(g0.y, g1.y);
        Btile32[(nb + 2) * 16 + r2] = pack_bf16(g0.z, g1.z);
        Btile32[(nb + 3) * 16 + r2] = pack_bf16(g0.w, g1.w);
      }
      __syncthreads();
      if (h == 0) afrag = frag_at(Atile32 + laneLo * 16 + laneHi * 8);
#pragma unroll
      for (int nt = 0; nt < 4; ++nt) {
        v16bf bfrag =
            frag_at(Btile32 + (wave * 64 + nt * 16 + laneLo) * 16 + laneHi * 8);
        acc[h * 4 + nt] = wmma_bf16(afrag, bfrag, acc[h * 4 + nt]);
      }
    }
  }

  // ---------- fused softmax over N=1024 ----------
  float rmax[8];
#pragma unroll
  for (int r = 0; r < 8; ++r) {
    float m = -INFINITY;
#pragma unroll
    for (int t = 0; t < 8; ++t) m = fmaxf(m, acc[t][r]);
    rmax[r] = redmax16(m);
  }
  if (laneLo == 0) {
#pragma unroll
    for (int r = 0; r < 8; ++r) red[wave * 16 + laneHi * 8 + r] = rmax[r];
  }
  __syncthreads();
  if (tid < 16) {
    float g = -INFINITY;
#pragma unroll
    for (int w = 0; w < 8; ++w) g = fmaxf(g, red[w * 16 + tid]);
    rowstat[tid] = g;
  }
  __syncthreads();
  float rowmax[8];
#pragma unroll
  for (int r = 0; r < 8; ++r) rowmax[r] = rowstat[laneHi * 8 + r];

  float rsum[8];
#pragma unroll
  for (int r = 0; r < 8; ++r) {
    float s = 0.0f;
#pragma unroll
    for (int t = 0; t < 8; ++t) {
      float e = __expf(acc[t][r] - rowmax[r]);
      acc[t][r] = e;
      s += e;
    }
    rsum[r] = redsum16(s);
  }
  if (laneLo == 0) {
#pragma unroll
    for (int r = 0; r < 8; ++r) red[wave * 16 + laneHi * 8 + r] = rsum[r];
  }
  __syncthreads();
  if (tid < 16) {
    float g = 0.0f;
#pragma unroll
    for (int w = 0; w < 8; ++w) g += red[w * 16 + tid];
    rowstat[tid] = g;
  }
  __syncthreads();
  float rinv[8];
#pragma unroll
  for (int r = 0; r < 8; ++r) rinv[r] = 1.0f / rowstat[laneHi * 8 + r];

#pragma unroll
  for (int t = 0; t < 8; ++t) {
    int n = ((t >= 4) ? 512 : 0) + wave * 64 + (t & 3) * 16 + laneLo;
#pragma unroll
    for (int r = 0; r < 8; ++r) {
      int m = r + 8 * laneHi;
      Aout[(size_t)(row0 + m) * T_DIM + n] = acc[t][r] * rinv[r];
    }
  }
}

// ====================================================================
// Kernel 2: per batch b: X_bar = W1 @ x[b]; gate; out = relu(X_tilde@W2 + B).
// LDS operand layouts (fragment order):
//   W1p32 [o*32 + d]  : A-operand, [m][k] pairs (K=64 padded)
//   xch32 [t*32 + f2] : B-operand, [n][k] pairs (K=f padded to 64)
//   W2ch32[n*32 + k2] : B-operand, [n][k] pairs (K=t chunk of 64)
//   Xt16  [m*64 + t]  : A-operand of 2nd GEMM, bf16 row-major
//   Ach   [f*64 + t]  : f32 gating chunk, loaded via async global->LDS
// ====================================================================
__global__ __launch_bounds__(256) void k_fused_out(const float* __restrict__ x,
                                                   const float* __restrict__ W1,
                                                   const float* __restrict__ W2,
                                                   const float* __restrict__ Bmat,
                                                   const float* __restrict__ lamp,
                                                   const float* __restrict__ Ain,
                                                   float* __restrict__ out) {
  __shared__ __align__(32) unsigned W1p32[128 * 32];   // 16 KB
  __shared__ __align__(32) unsigned xch32[64 * 32];    //  8 KB
  __shared__ __align__(32) unsigned W2ch32[64 * 32];   //  8 KB
  __shared__ __align__(32) float    Ach[D1_IN * 64];   // 10 KB
  __shared__ __align__(32) __bf16   Xt16[128 * 64];    // 16 KB

  const int tid    = threadIdx.x;
  const int wave   = tid >> 5;
  const int lane   = tid & 31;
  const int laneLo = lane & 15;
  const int laneHi = lane >> 4;
  const int b      = blockIdx.x;

  // ---- stage padded W1 (128 x 64 bf16) once, fragment-order ----
#pragma unroll
  for (int it = 0; it < 16; ++it) {
    int i  = it * 256 + tid;
    int o  = i >> 5;
    int f  = (i & 31) * 2;
    float a  = (o < D1_OUT && f < D1_IN) ? W1[o * D1_IN + f] : 0.0f;
    float bb = (o < D1_OUT && f + 1 < D1_IN) ? W1[o * D1_IN + f + 1] : 0.0f;
    int d = f >> 1;
    W1p32[o * 32 + d] = pack_bf16(a, bb);
  }

  float lam = fminf(fmaxf(lamp[0], 0.0f), 1.0f);
  float oml = 1.0f - lam;

  v8f accO[4];
#pragma unroll
  for (int nt = 0; nt < 4; ++nt) accO[nt] = zero8();

  for (int t0 = 0; t0 < T_DIM; t0 += 64) {
    __syncthreads();
    // ---- x chunk (K=f padded to 64, N=t 64), [t][f-pair] dwords ----
#pragma unroll
    for (int it = 0; it < 8; ++it) {
      int i  = it * 256 + tid;
      int f2 = i >> 6;
      int t  = i & 63;
      int f  = f2 * 2;
      float a  = (f < D1_IN)
                     ? x[((size_t)b * D1_IN + f) * T_DIM + t0 + t] : 0.0f;
      float bb = (f + 1 < D1_IN)
                     ? x[((size_t)b * D1_IN + f + 1) * T_DIM + t0 + t] : 0.0f;
      xch32[t * 32 + f2] = pack_bf16(a, bb);
    }
    // ---- W2 chunk (K=t 64, N=64), [n][k-pair] dwords ----
#pragma unroll
    for (int it = 0; it < 8; ++it) {
      int i  = it * 256 + tid;
      int k2 = i >> 6;
      int n  = i & 63;
      float a  = W2[(size_t)(t0 + 2 * k2) * D2_OUT + n];
      float bb = W2[(size_t)(t0 + 2 * k2 + 1) * D2_OUT + n];
      W2ch32[n * 32 + k2] = pack_bf16(a, bb);
    }
    // ---- A gating chunk: async global -> LDS (raw f32, ASYNCcnt-tracked) ----
#pragma unroll
    for (int it = 0; it < 10; ++it) {
      int i = it * 256 + tid;
      int f = i >> 6;
      int t = i & 63;
      unsigned lds_off = (unsigned)(uintptr_t)(&Ach[i]);
      unsigned long long gaddr =
          (unsigned long long)(uintptr_t)(Ain + ((size_t)b * D1_IN + f) * T_DIM +
                                          t0 + t);
      asm volatile("global_load_async_to_lds_b32 %0, %1, off"
                   :: "v"(lds_off), "v"(gaddr)
                   : "memory");
    }
    asm volatile("s_wait_asynccnt 0" ::: "memory");
    __syncthreads();

    // ---- X_bar chunk: rows [wave*16, wave*16+16) x 64 cols ----
    v8f accx[4];
#pragma unroll
    for (int nt = 0; nt < 4; ++nt) accx[nt] = zero8();
#pragma unroll
    for (int kk = 0; kk < 64; kk += 32) {
      v16bf af =
          frag_at(W1p32 + (wave * 16 + laneLo) * 32 + (kk >> 1) + laneHi * 8);
#pragma unroll
      for (int nt = 0; nt < 4; ++nt) {
        v16bf bf_ =
            frag_at(xch32 + (nt * 16 + laneLo) * 32 + (kk >> 1) + laneHi * 8);
        accx[nt] = wmma_bf16(af, bf_, accx[nt]);
      }
    }

    // ---- gate -> X_tilde in LDS (row-major bf16, A-operand order) ----
#pragma unroll
    for (int nt = 0; nt < 4; ++nt) {
#pragma unroll
      for (int r = 0; r < 8; ++r) {
        int m = wave * 16 + r + 8 * laneHi;   // 0..127 (rows >=120 are zero)
        int t = nt * 16 + laneLo;
        int f = m % D1_IN;
        float g = lam * Ach[f * 64 + t] + oml;
        int ts = t;
        Xt16[m * 64 + ts] = (__bf16)(accx[nt][r] * g);
      }
    }
    __syncthreads();

    // ---- out += X_tilde @ W2_chunk ----
#pragma unroll
    for (int kk = 0; kk < 64; kk += 32) {
      v16bf af = *(const v16bf*)(Xt16 + (wave * 16 + laneLo) * 64 + kk +
                                 laneHi * 16);
#pragma unroll
      for (int nt = 0; nt < 4; ++nt) {
        v16bf bf_ =
            frag_at(W2ch32 + (nt * 16 + laneLo) * 32 + (kk >> 1) + laneHi * 8);
        accO[nt] = wmma_bf16(af, bf_, accO[nt]);
      }
    }
  }

  // ---- epilogue: + B, relu, store ----
#pragma unroll
  for (int nt = 0; nt < 4; ++nt) {
#pragma unroll
    for (int r = 0; r < 8; ++r) {
      int m = wave * 16 + r + 8 * laneHi;
      int n = nt * 16 + laneLo;
      if (m < D1_OUT) {
        float v = accO[nt][r] + Bmat[m * D2_OUT + n];
        out[((size_t)b * D1_OUT + m) * D2_OUT + n] = fmaxf(v, 0.0f);
      }
    }
  }
}

// ====================================================================
extern "C" void kernel_launch(void* const* d_in, const int* in_sizes, int n_in,
                              void* d_out, int out_size, void* d_ws, size_t ws_size,
                              hipStream_t stream) {
  (void)in_sizes; (void)n_in; (void)out_size; (void)d_ws; (void)ws_size;
  const float* x    = (const float*)d_in[0];
  const float* W1   = (const float*)d_in[1];
  const float* W    = (const float*)d_in[2];
  const float* W2   = (const float*)d_in[3];
  const float* Bmat = (const float*)d_in[4];
  const float* lamp = (const float*)d_in[5];

  float* out = (float*)d_out;
  float* A   = out + (size_t)B_SZ * D1_OUT * D2_OUT;  // second tuple element

  k_attn_softmax<<<(B_SZ * D1_IN) / 16, 256, 0, stream>>>(x, W, A);
  k_fused_out<<<B_SZ, 256, 0, stream>>>(x, W1, W2, Bmat, lamp, A, out);
}
